// HiViT_16647293239785
// MI455X (gfx1250) — matrix operand order repaired and checked by reading, not compiled, not run on hardware
//
#include <hip/hip_runtime.h>

// ---------------------------------------------------------------------------
// Types for CDNA5 WMMA (gfx1250, wave32)
// ---------------------------------------------------------------------------
typedef __bf16 bf16_t;
typedef bf16_t v16bf __attribute__((ext_vector_type(16)));
typedef float  v8f   __attribute__((ext_vector_type(8)));

__device__ __forceinline__ unsigned short f2bf(float f) {
  unsigned int u = __float_as_uint(f);
  u += 0x7FFFu + ((u >> 16) & 1u);          // round-to-nearest-even
  return (unsigned short)(u >> 16);
}

union FragBF { v16bf v; uint4 q[2]; };

// LDS B-tile: 64 rows x 64B of data, padded to 80B/row so the 16 rows a wave
// reads per fragment stride 20 banks (mod 64) instead of 16 (4-way conflict).
#define BROW 80
#define BBUF (64 * BROW)

// ---------------------------------------------------------------------------
// WMMA GEMM:  C[M,N] = A[M,K] (bf16, row-major, ld=lda)
//                    x B[N,K]^T (bf16 weights, row-major [N][K], ld=ldb)
// 256 threads = 8 waves; block tile 256x64; wave w owns 32 rows
// (m0 = blk.x*256 + w*32) with 8 accumulators over the 64 columns.
// B chunk (64x32) staged to LDS with async global->LDS copies (ASYNCcnt),
// double buffered; A fragments register-prefetched one chunk ahead with
// explicit ping-pong (K/32 is even for every GEMM in this model, so the
// 2x-unrolled pipeline needs no tail and no cross-iteration register copies).
// Batched via blockIdx.z -> (bb, hh) with independent strides.
// ---------------------------------------------------------------------------
template<bool BIAS, bool GELU, bool RESID, bool OUTBF16>
__global__ __launch_bounds__(256) void gemm_wmma(
    const bf16_t* __restrict__ A, int lda, long sAb, long sAh,
    const bf16_t* __restrict__ Bm, int ldb, long sBb, long sBh,
    void* __restrict__ Cv, int ldc, long sCb, long sCh,
    const float* __restrict__ bias,
    const float* __restrict__ resid,
    int K, int H)
{
  __shared__ unsigned char lsB[2][BBUF] __attribute__((aligned(16)));

  const int z  = blockIdx.z;
  const int bb = z / H, hh = z - bb * H;
  A  += (long)bb * sAb + (long)hh * sAh;
  Bm += (long)bb * sBb + (long)hh * sBh;
  const long coff = (long)bb * sCb + (long)hh * sCh;

  const int tid  = threadIdx.x;
  const int wave = tid >> 5;
  const int lane = tid & 31;
  const int half = lane >> 4;
  const int idx  = lane & 15;
  const int m0 = blockIdx.x * 256 + wave * 32;
  const int n0 = blockIdx.y * 64;

  // --- LDS staging setup: thread t copies 16B: row = t/4, seg = t%4 ---
  const int srow = tid >> 2, sseg = tid & 3;
  const bf16_t* gstage = Bm + (long)(n0 + srow) * ldb + sseg * 8;
  unsigned ldsbase;
  {
    auto p3 = (__attribute__((address_space(3))) unsigned char*)(&lsB[0][0]);
    ldsbase = (unsigned)(unsigned long long)p3;
  }
  const unsigned ldst = ldsbase + (unsigned)(srow * BROW + sseg * 16);

  auto stage = [&](int buf, int kb) {
    unsigned dst = ldst + (unsigned)(buf * BBUF);
    const bf16_t* src = gstage + kb;
    asm volatile("global_load_async_to_lds_b128 %0, %1, off"
                 :: "v"(dst), "v"(src) : "memory");
  };

  // A fragment rows: lane idx -> M; K striping: lanes0-15 K 0..7,16..23,
  // lanes16-31 K 8..15,24..31 (16-bit A layout, ISA 7.12.2). Two row groups.
  const bf16_t* arow0 = A + (long)(m0 + idx) * lda + half * 8;
  const bf16_t* arow1 = arow0 + 16 * lda;

  auto loadA = [&](FragBF* f, int c) {
    const int kb = c * 32;
    f[0].q[0] = *(const uint4*)(arow0 + kb);
    f[0].q[1] = *(const uint4*)(arow0 + kb + 16);
    f[1].q[0] = *(const uint4*)(arow1 + kb);
    f[1].q[1] = *(const uint4*)(arow1 + kb + 16);
  };

  v8f acc[2][4] = {};

  auto compute = [&](const FragBF* f, int buf) {
    // B fragments from LDS: lane idx -> column n; lanes0-15 K 0..7,16..23,
    // lanes16-31 K 8..15,24..31 of each padded 80B row.
    const unsigned char* bbase =
        (const unsigned char*)lsB + buf * BBUF + half * 16;
#pragma unroll
    for (int j = 0; j < 4; ++j) {
      FragBF bf;
      const unsigned char* bp = bbase + (j * 16 + idx) * BROW;
      bf.q[0] = *(const uint4*)(bp);
      bf.q[1] = *(const uint4*)(bp + 32);
      acc[0][j] = __builtin_amdgcn_wmma_f32_16x16x32_bf16(
          false, f[0].v, false, bf.v, (short)0, acc[0][j], false, false);
      acc[1][j] = __builtin_amdgcn_wmma_f32_16x16x32_bf16(
          false, f[1].v, false, bf.v, (short)0, acc[1][j], false, false);
    }
  };

  // ---- pipeline prologue: stage chunk 0 -> buf0, prefetch A chunk 0 ----
  FragBF aA[2], aB[2];
  stage(0, 0);
  loadA(aA, 0);
  asm volatile("s_wait_asynccnt 0x0" ::: "memory");
  __syncthreads();

  const int nchunk = K >> 5;                 // even for all GEMMs here
  for (int c = 0; c < nchunk; c += 2) {
    // chunk c lives in buf0, chunk c+1 in buf1
    stage(1, (c + 1) * 32);
    loadA(aB, c + 1);
    compute(aA, 0);
    asm volatile("s_wait_asynccnt 0x0" ::: "memory");
    __syncthreads();                          // buf1 ready; buf0 free
    if (c + 2 < nchunk) {
      stage(0, (c + 2) * 32);
      loadA(aA, c + 2);
    }
    compute(aB, 1);
    asm volatile("s_wait_asynccnt 0x0" ::: "memory");
    __syncthreads();                          // buf0 ready; buf1 free
  }

  // C layout: vgpr r -> M = m0 + g*16 + half*8 + r ; lane idx -> N
#pragma unroll
  for (int g = 0; g < 2; ++g) {
#pragma unroll
    for (int j = 0; j < 4; ++j) {
      const int n = n0 + j * 16 + idx;
      float bv = 0.f;
      if (BIAS) bv = bias[n];
#pragma unroll
      for (int r = 0; r < 8; ++r) {
        const int m = m0 + g * 16 + half * 8 + r;
        float v = acc[g][j][r] + bv;
        if (GELU) v = 0.5f * v * (1.f + erff(v * 0.70710678118654752f));
        if (RESID) v += resid[(long)m * ldc + n];
        const long off = coff + (long)m * ldc + n;
        if (OUTBF16) ((unsigned short*)Cv)[off] = f2bf(v);
        else         ((float*)Cv)[off]          = v;
      }
    }
  }
}

// ---------------------------------------------------------------------------
// LayerNorm over last dim C (fp32 in, bf16 or fp32 out). 1 block / row.
// ---------------------------------------------------------------------------
template<bool OUTBF16>
__global__ __launch_bounds__(128) void ln_kernel(
    const float* __restrict__ x, const float* __restrict__ w,
    const float* __restrict__ b, void* __restrict__ out, int C)
{
  const long row = blockIdx.x;
  const float* xr = x + row * C;
  float s = 0.f, s2 = 0.f;
  for (int c = threadIdx.x; c < C; c += 128) {
    float v = xr[c]; s += v; s2 += v * v;
  }
  __shared__ float rs[128], rq[128];
  rs[threadIdx.x] = s; rq[threadIdx.x] = s2;
  __syncthreads();
  for (int off = 64; off > 0; off >>= 1) {
    if (threadIdx.x < off) { rs[threadIdx.x] += rs[threadIdx.x + off];
                             rq[threadIdx.x] += rq[threadIdx.x + off]; }
    __syncthreads();
  }
  const float mean = rs[0] / C;
  const float var  = rq[0] / C - mean * mean;
  const float inv  = rsqrtf(var + 1e-5f);
  for (int c = threadIdx.x; c < C; c += 128) {
    float v = (xr[c] - mean) * inv * w[c] + b[c];
    if (OUTBF16) ((unsigned short*)out)[row * C + c] = f2bf(v);
    else         ((float*)out)[row * C + c]          = v;
  }
}

// ---------------------------------------------------------------------------
// PatchEmbed: 4x4/stride4 conv + HiViT token reorder. 1 block / token.
// out[b*4096 + t][oc],  t = ((hp*16+wp)*4 + r)*4 + c  (row=hp*4+r, col=wp*4+c)
// ---------------------------------------------------------------------------
__global__ __launch_bounds__(128) void patch_embed_kernel(
    const float* __restrict__ img, const float* __restrict__ w,
    const float* __restrict__ bias, float* __restrict__ out)
{
  const int tk = blockIdx.x;
  const int b = tk >> 12, t = tk & 4095;
  const int c2 = t & 3, r2 = (t >> 2) & 3, wp = (t >> 4) & 15, hp = t >> 8;
  const int row = hp * 4 + r2, col = wp * 4 + c2;
  __shared__ float patch[48];
  if (threadIdx.x < 48) {
    const int ci = threadIdx.x >> 4, kh = (threadIdx.x >> 2) & 3, kw = threadIdx.x & 3;
    patch[threadIdx.x] =
        img[(((long)b * 3 + ci) * 256 + row * 4 + kh) * 256 + col * 4 + kw];
  }
  __syncthreads();
  const int oc = threadIdx.x;
  const float* wr = w + oc * 48;
  float acc = bias[oc];
#pragma unroll
  for (int i = 0; i < 48; ++i) acc += patch[i] * wr[i];
  out[(long)tk * 128 + oc] = acc;
}

// ---------------------------------------------------------------------------
// Patch-merge gather: [B,256,s,s,C] -> [B,256,s/2,s/2,4C] (concat 2x2 quads)
// ---------------------------------------------------------------------------
__global__ void merge_gather_kernel(const float* __restrict__ in,
                                    float* __restrict__ out,
                                    int C, int s, int total)
{
  int e = blockIdx.x * 256 + threadIdx.x;
  if (e >= total) return;
  const int half = s >> 1;
  int r = e;
  const int c = r % C; r /= C;
  const int q = r & 3;  r >>= 2;
  const int j = r % half; r /= half;
  const int i = r % half; r /= half;
  const int p = r % 256;
  const int b = r / 256;
  const int qr = q & 1, qc = q >> 1;           // q: (0,0),(1,0),(0,1),(1,1)
  const int tok = p * s * s + (2 * i + qr) * s + (2 * j + qc);
  out[e] = in[((long)b * 256 * s * s + tok) * C + c];
}

__global__ void add_pos_kernel(float* __restrict__ x,
                               const float* __restrict__ pos, int total)
{
  int e = blockIdx.x * 256 + threadIdx.x;
  if (e < total) x[e] += pos[e % (256 * 512)];
}

// v slice of qkv -> vT[b,h,d,m] (bf16) so it is [N][K] for the P@v GEMM
__global__ void vtrans_kernel(const unsigned short* __restrict__ qkv,
                              unsigned short* __restrict__ vt)
{
  const int e = blockIdx.x * 256 + threadIdx.x;  // 16*8*64*256 = 2097152
  const int m = e & 255, d = (e >> 8) & 63, h = (e >> 14) & 7, b = e >> 17;
  vt[e] = qkv[((long)(b * 256 + m)) * 1536 + 1024 + h * 64 + d];
}

// scores row softmax with relative-position bias gather; out bf16 probs
__global__ __launch_bounds__(256) void softmax_bias_kernel(
    const float* __restrict__ S, unsigned short* __restrict__ P,
    const float* __restrict__ rpb, const int* __restrict__ rpe, float scale)
{
  const int n = blockIdx.x, h = blockIdx.y, b = blockIdx.z, m = threadIdx.x;
  const long base = (((long)b * 8 + h) * 256 + n) * 256;
  float v = S[base + m] * scale + rpb[(long)rpe[n * 256 + m] * 8 + h];
  __shared__ float red[256];
  red[m] = v; __syncthreads();
  for (int off = 128; off > 0; off >>= 1) {
    if (m < off) red[m] = fmaxf(red[m], red[m + off]);
    __syncthreads();
  }
  const float mx = red[0]; __syncthreads();
  const float e = expf(v - mx);
  red[m] = e; __syncthreads();
  for (int off = 128; off > 0; off >>= 1) {
    if (m < off) red[m] += red[m + off];
    __syncthreads();
  }
  P[base + m] = f2bf(e / red[0]);
}

__global__ void cvt_bf16_kernel(const float* __restrict__ in,
                                unsigned short* __restrict__ out, long n)
{
  long e = (long)blockIdx.x * 256 + threadIdx.x;
  if (e < n) out[e] = f2bf(in[e]);
}

// ---------------------------------------------------------------------------
#define GEMM(BIAS, GELU, RESID, OBF, gx, gy, gz, Ap, lda, sAb, sAh, Bp, ldb, \
             sBb, sBh, Cp, ldc, sCb, sCh, biasp, residp, Kk, Hh)              \
  gemm_wmma<BIAS, GELU, RESID, OBF><<<dim3(gx, gy, gz), 256, 0, stream>>>(    \
      (const bf16_t*)(Ap), lda, sAb, sAh, (const bf16_t*)(Bp), ldb, sBb, sBh, \
      (void*)(Cp), ldc, sCb, sCh, biasp, residp, Kk, Hh)

extern "C" void kernel_launch(void* const* d_in, const int* in_sizes, int n_in,
                              void* d_out, int out_size, void* d_ws,
                              size_t ws_size, hipStream_t stream)
{
  (void)in_sizes; (void)n_in; (void)out_size; (void)ws_size;

  const float* x_img    = (const float*)d_in[0];
  const float* patch_w  = (const float*)d_in[1];
  const float* patch_b  = (const float*)d_in[2];
  const float* pnorm_w  = (const float*)d_in[3];
  const float* pnorm_b  = (const float*)d_in[4];
  const float* s0_n2w   = (const float*)d_in[5];
  const float* s0_n2b   = (const float*)d_in[6];
  const float* s0_fc1w  = (const float*)d_in[7];
  const float* s0_fc1b  = (const float*)d_in[8];
  const float* s0_fc2w  = (const float*)d_in[9];
  const float* s0_fc2b  = (const float*)d_in[10];
  const float* m0_nw    = (const float*)d_in[11];
  const float* m0_nb    = (const float*)d_in[12];
  const float* m0_rw    = (const float*)d_in[13];
  const float* s1_n2w   = (const float*)d_in[14];
  const float* s1_n2b   = (const float*)d_in[15];
  const float* s1_fc1w  = (const float*)d_in[16];
  const float* s1_fc1b  = (const float*)d_in[17];
  const float* s1_fc2w  = (const float*)d_in[18];
  const float* s1_fc2b  = (const float*)d_in[19];
  const float* m1_nw    = (const float*)d_in[20];
  const float* m1_nb    = (const float*)d_in[21];
  const float* m1_rw    = (const float*)d_in[22];
  const float* pos_emb  = (const float*)d_in[23];
  const float* mb_n1w   = (const float*)d_in[24];
  const float* mb_n1b   = (const float*)d_in[25];
  const float* mb_qkvw  = (const float*)d_in[26];
  const float* mb_qkvb  = (const float*)d_in[27];
  const float* mb_rpb   = (const float*)d_in[28];
  const float* mb_projw = (const float*)d_in[29];
  const float* mb_projb = (const float*)d_in[30];
  const float* mb_n2w   = (const float*)d_in[31];
  const float* mb_n2b   = (const float*)d_in[32];
  const float* mb_fc1w  = (const float*)d_in[33];
  const float* mb_fc1b  = (const float*)d_in[34];
  const float* mb_fc2w  = (const float*)d_in[35];
  const float* mb_fc2b  = (const float*)d_in[36];
  const float* fnorm_w  = (const float*)d_in[37];
  const float* fnorm_b  = (const float*)d_in[38];
  const int*   rpe_idx  = (const int*)d_in[39];

  // --- workspace carve ---
  unsigned char* W = (unsigned char*)d_ws;
  size_t off = 0;
  auto take = [&](size_t bytes) -> void* {
    void* p = W + off;
    off += (bytes + 255) & ~(size_t)255;
    return p;
  };
  unsigned short* w_qkv  = (unsigned short*)take(20L * 1536 * 512 * 2);
  unsigned short* w_proj = (unsigned short*)take(20L * 512 * 512 * 2);
  unsigned short* w_f1   = (unsigned short*)take(20L * 2048 * 512 * 2);
  unsigned short* w_f2   = (unsigned short*)take(20L * 512 * 2048 * 2);
  unsigned short* w_s0f1 = (unsigned short*)take(8L * 384 * 128 * 2);
  unsigned short* w_s0f2 = (unsigned short*)take(8L * 128 * 384 * 2);
  unsigned short* w_s1f1 = (unsigned short*)take(8L * 768 * 256 * 2);
  unsigned short* w_s1f2 = (unsigned short*)take(8L * 256 * 768 * 2);
  unsigned short* w_m0   = (unsigned short*)take(256L * 512 * 2);
  unsigned short* w_m1   = (unsigned short*)take(512L * 1024 * 2);
  float*          xbuf   = (float*)take(8388608L * 4);   // residual stream
  float*          ybuf   = (float*)take(8388608L * 4);   // conv / merge staging
  unsigned short* hbuf   = (unsigned short*)take(8388608L * 2); // post-LN / O
  unsigned short* midbuf = (unsigned short*)take(25165824L * 2); // MLP hidden
  unsigned short* qkvbuf = (unsigned short*)take(6291456L * 2);
  float*          sbuf   = (float*)take(8388608L * 4);   // attn scores f32
  unsigned short* pbuf   = (unsigned short*)take(8388608L * 2); // probs bf16
  unsigned short* vtbuf  = (unsigned short*)take(2097152L * 2); // v^T bf16

  auto cvt = [&](const float* src, unsigned short* dst, long n) {
    cvt_bf16_kernel<<<dim3((unsigned)((n + 255) / 256)), 256, 0, stream>>>(src, dst, n);
  };
  cvt(mb_qkvw,  w_qkv,  20L * 1536 * 512);
  cvt(mb_projw, w_proj, 20L * 512 * 512);
  cvt(mb_fc1w,  w_f1,   20L * 2048 * 512);
  cvt(mb_fc2w,  w_f2,   20L * 512 * 2048);
  cvt(s0_fc1w,  w_s0f1, 8L * 384 * 128);
  cvt(s0_fc2w,  w_s0f2, 8L * 128 * 384);
  cvt(s1_fc1w,  w_s1f1, 8L * 768 * 256);
  cvt(s1_fc2w,  w_s1f2, 8L * 256 * 768);
  cvt(m0_rw,    w_m0,   256L * 512);
  cvt(m1_rw,    w_m1,   512L * 1024);

  // --- patch embed + pnorm ---  tokens: 16*4096 rows of 128
  patch_embed_kernel<<<65536, 128, 0, stream>>>(x_img, patch_w, patch_b, ybuf);
  ln_kernel<false><<<65536, 128, 0, stream>>>(ybuf, pnorm_w, pnorm_b, xbuf, 128);

  // --- stem stage 0: 8 MLP blocks, dim 128 -> hidden 384 ---
  for (int l = 0; l < 8; ++l) {
    ln_kernel<true><<<65536, 128, 0, stream>>>(xbuf, s0_n2w + l * 128,
                                               s0_n2b + l * 128, hbuf, 128);
    GEMM(true, true, false, true, 256, 6, 1,
         hbuf, 128, 0L, 0L, w_s0f1 + (long)l * 384 * 128, 128, 0L, 0L,
         midbuf, 384, 0L, 0L, s0_fc1b + l * 384, (const float*)nullptr, 128, 1);
    GEMM(true, false, true, false, 256, 2, 1,
         midbuf, 384, 0L, 0L, w_s0f2 + (long)l * 128 * 384, 384, 0L, 0L,
         xbuf, 128, 0L, 0L, s0_fc2b + l * 128, xbuf, 384, 1);
  }

  // --- merge 0: [16,256,4,4,128] -> [16,1024,512] -> GEMM -> dim 256 ---
  merge_gather_kernel<<<8388608 / 256, 256, 0, stream>>>(xbuf, ybuf, 128, 4, 8388608);
  ln_kernel<true><<<16384, 128, 0, stream>>>(ybuf, m0_nw, m0_nb, hbuf, 512);
  GEMM(false, false, false, false, 64, 4, 1,
       hbuf, 512, 0L, 0L, w_m0, 512, 0L, 0L,
       xbuf, 256, 0L, 0L, (const float*)nullptr, (const float*)nullptr, 512, 1);

  // --- stem stage 1: 8 MLP blocks, dim 256 -> hidden 768 ---
  for (int l = 0; l < 8; ++l) {
    ln_kernel<true><<<16384, 128, 0, stream>>>(xbuf, s1_n2w + l * 256,
                                               s1_n2b + l * 256, hbuf, 256);
    GEMM(true, true, false, true, 64, 12, 1,
         hbuf, 256, 0L, 0L, w_s1f1 + (long)l * 768 * 256, 256, 0L, 0L,
         midbuf, 768, 0L, 0L, s1_fc1b + l * 768, (const float*)nullptr, 256, 1);
    GEMM(true, false, true, false, 64, 4, 1,
         midbuf, 768, 0L, 0L, w_s1f2 + (long)l * 256 * 768, 768, 0L, 0L,
         xbuf, 256, 0L, 0L, s1_fc2b + l * 256, xbuf, 768, 1);
  }

  // --- merge 1: [16,256,2,2,256] -> [16,256,1024] -> GEMM -> dim 512 ---
  merge_gather_kernel<<<4194304 / 256, 256, 0, stream>>>(xbuf, ybuf, 256, 2, 4194304);
  ln_kernel<true><<<4096, 128, 0, stream>>>(ybuf, m1_nw, m1_nb, hbuf, 1024);
  GEMM(false, false, false, false, 16, 8, 1,
       hbuf, 1024, 0L, 0L, w_m1, 1024, 0L, 0L,
       xbuf, 512, 0L, 0L, (const float*)nullptr, (const float*)nullptr, 1024, 1);
  add_pos_kernel<<<8192, 256, 0, stream>>>(xbuf, pos_emb, 2097152);

  // --- 20 main attention blocks (B=16, N=256, C=512, H=8, Dh=64) ---
  for (int l = 0; l < 20; ++l) {
    ln_kernel<true><<<4096, 128, 0, stream>>>(xbuf, mb_n1w + l * 512,
                                              mb_n1b + l * 512, hbuf, 512);
    // qkv: [4096,512] x [1536,512]^T -> qkvbuf (bf16)
    GEMM(true, false, false, true, 16, 24, 1,
         hbuf, 512, 0L, 0L, w_qkv + (long)l * 1536 * 512, 512, 0L, 0L,
         qkvbuf, 1536, 0L, 0L, mb_qkvb + l * 1536, (const float*)nullptr, 512, 1);
    vtrans_kernel<<<8192, 256, 0, stream>>>(qkvbuf, vtbuf);
    // scores: per (b,h) 256x256 = q[256,64] x k[256,64]^T  (f32 out)
    GEMM(false, false, false, false, 1, 4, 128,
         qkvbuf, 1536, 256L * 1536, 64L,
         qkvbuf + 512, 1536, 256L * 1536, 64L,
         sbuf, 256, 8L * 65536, 65536L,
         (const float*)nullptr, (const float*)nullptr, 64, 8);
    softmax_bias_kernel<<<dim3(256, 8, 16), 256, 0, stream>>>(
        sbuf, pbuf, mb_rpb + (long)l * 961 * 8, rpe_idx, 0.125f);
    // O = P x v : per (b,h) [256,256] x vT[64,256]^T -> hbuf[b*256+m][h*64+d]
    GEMM(false, false, false, true, 1, 1, 128,
         pbuf, 256, 8L * 65536, 65536L,
         vtbuf, 256, 8L * 16384, 16384L,
         hbuf, 512, 256L * 512, 64L,
         (const float*)nullptr, (const float*)nullptr, 256, 8);
    // proj + residual
    GEMM(true, false, true, false, 16, 8, 1,
         hbuf, 512, 0L, 0L, w_proj + (long)l * 512 * 512, 512, 0L, 0L,
         xbuf, 512, 0L, 0L, mb_projb + l * 512, xbuf, 512, 1);
    // MLP
    ln_kernel<true><<<4096, 128, 0, stream>>>(xbuf, mb_n2w + l * 512,
                                              mb_n2b + l * 512, hbuf, 512);
    GEMM(true, true, false, true, 16, 32, 1,
         hbuf, 512, 0L, 0L, w_f1 + (long)l * 2048 * 512, 512, 0L, 0L,
         midbuf, 2048, 0L, 0L, mb_fc1b + l * 2048, (const float*)nullptr, 512, 1);
    GEMM(true, false, true, false, 16, 8, 1,
         midbuf, 2048, 0L, 0L, w_f2 + (long)l * 512 * 2048, 2048, 0L, 0L,
         xbuf, 512, 0L, 0L, mb_fc2b + l * 512, xbuf, 2048, 1);
  }

  // --- final LayerNorm -> fp32 output [16,256,512] ---
  ln_kernel<false><<<4096, 128, 0, stream>>>(xbuf, fnorm_w, fnorm_b, d_out, 512);
}
